// SingleLayerMoE_42700564857409
// MI455X (gfx1250) — matrix operand order repaired and checked by reading, not compiled, not run on hardware
//
#include <hip/hip_runtime.h>
#include <hip/hip_bf16.h>

#define NTOK 8192
#define DIMD 1024
#define HID  4096
#define NEXP 8

typedef __attribute__((ext_vector_type(16))) __bf16 v16bf;
typedef __attribute__((ext_vector_type(8)))  float  v8f;

union FragAB { v16bf v; uint4 u[2]; };

__device__ __forceinline__ unsigned short f2bf(float f) {
  unsigned int u = __float_as_uint(f);
  u += 0x7FFFu + ((u >> 16) & 1u);   // round-to-nearest-even
  return (unsigned short)(u >> 16);
}

__device__ __forceinline__ v8f bf_wmma(FragAB a, FragAB b, v8f c) {
  return __builtin_amdgcn_wmma_f32_16x16x32_bf16(false, a.v, false, b.v,
                                                 (short)0, c, false, false);
}

// ---------------------------------------------------------------------------
// Zero output + stats
// ---------------------------------------------------------------------------
__global__ void zero_kernel(float* __restrict__ out, int n_out, float* __restrict__ stats) {
  int i = blockIdx.x * blockDim.x + threadIdx.x;
  if (i < n_out) out[i] = 0.0f;
  if (i < 9) stats[i] = 0.0f;
}

// ---------------------------------------------------------------------------
// x fp32 -> bf16 (row-major kept; rows are the WMMA A K-dimension)
// ---------------------------------------------------------------------------
__global__ void convx_kernel(const float* __restrict__ x, unsigned short* __restrict__ xb) {
  int i = blockIdx.x * blockDim.x + threadIdx.x;   // one float4
  float4 v = ((const float4*)x)[i];
  ushort4 o;
  o.x = f2bf(v.x); o.y = f2bf(v.y); o.z = f2bf(v.z); o.w = f2bf(v.w);
  ((ushort4*)xb)[i] = o;
}

// ---------------------------------------------------------------------------
// Per-expert transpose + fp32->bf16: src [E][R][C] f32 -> dst [E][C][R] bf16
// So the WMMA B operand is contiguous along K per output column.
// ---------------------------------------------------------------------------
__global__ void transpose_kernel(const float* __restrict__ src,
                                 unsigned short* __restrict__ dst,
                                 int R, int C) {
  __shared__ float tile[32][33];
  const int z = blockIdx.z;
  src += (size_t)z * R * C;
  dst += (size_t)z * R * C;
  const int c  = blockIdx.x * 32 + threadIdx.x;
  const int rb = blockIdx.y * 32;
#pragma unroll
  for (int i = 0; i < 4; ++i) {
    int r = rb + threadIdx.y + i * 8;
    tile[threadIdx.y + i * 8][threadIdx.x] = src[(size_t)r * C + c];
  }
  __syncthreads();
  const int orr = rb + threadIdx.x;
#pragma unroll
  for (int i = 0; i < 4; ++i) {
    int oc = blockIdx.x * 32 + threadIdx.y + i * 8;
    dst[(size_t)oc * R + orr] = f2bf(tile[threadIdx.x][threadIdx.y + i * 8]);
  }
}

// ---------------------------------------------------------------------------
// Router: per-token logits, top-2 + softmax gates, entropy/importance stats
// ---------------------------------------------------------------------------
__global__ void router_kernel(const float* __restrict__ x,
                              const float* __restrict__ rw,
                              const float* __restrict__ rbias,
                              unsigned int* __restrict__ tops,
                              float2* __restrict__ gates2,
                              float* __restrict__ stats) {
  __shared__ float s_red[9];
  const int tid = threadIdx.x;
  if (tid < 9) s_red[tid] = 0.0f;
  __syncthreads();

  const int t = blockIdx.x * blockDim.x + tid;
  float acc[8];
#pragma unroll
  for (int e = 0; e < 8; ++e) acc[e] = rbias[e];

  const float* xr = x + (size_t)t * DIMD;
  for (int d = 0; d < DIMD; ++d) {
    float xv = xr[d];
    const float4* w4 = (const float4*)(rw + d * 8);
    float4 a = w4[0], b = w4[1];
    acc[0] = fmaf(xv, a.x, acc[0]); acc[1] = fmaf(xv, a.y, acc[1]);
    acc[2] = fmaf(xv, a.z, acc[2]); acc[3] = fmaf(xv, a.w, acc[3]);
    acc[4] = fmaf(xv, b.x, acc[4]); acc[5] = fmaf(xv, b.y, acc[5]);
    acc[6] = fmaf(xv, b.z, acc[6]); acc[7] = fmaf(xv, b.w, acc[7]);
  }

  // top-2 (ties -> lowest index, matching lax.top_k)
  float best = -1e30f, sec = -1e30f; int bi = 0, si = 0;
#pragma unroll
  for (int e = 0; e < 8; ++e) {
    float l = acc[e];
    if (l > best) { sec = best; si = bi; best = l; bi = e; }
    else if (l > sec) { sec = l; si = e; }
  }
  float g0 = 1.0f / (1.0f + expf(sec - best));
  float g1 = 1.0f - g0;
  tops[t]   = (unsigned)bi | ((unsigned)si << 16);
  gates2[t] = make_float2(g0, g1);

  // full softmax stats
  float s = 0.0f, p[8];
#pragma unroll
  for (int e = 0; e < 8; ++e) { p[e] = expf(acc[e] - best); s += p[e]; }
  float inv = 1.0f / s, ent = 0.0f;
#pragma unroll
  for (int e = 0; e < 8; ++e) {
    float pe = p[e] * inv;
    ent -= pe * logf(fmaxf(pe, 1e-8f));
    atomicAdd(&s_red[1 + e], pe);
  }
  atomicAdd(&s_red[0], ent);
  __syncthreads();
  if (tid < 9) atomicAdd(&stats[tid], s_red[tid]);
}

// ---------------------------------------------------------------------------
// Deterministic per-expert token lists: one wave32 per expert, ballot scan.
// ---------------------------------------------------------------------------
__global__ void build_kernel(const unsigned int* __restrict__ tops,
                             const float2* __restrict__ gates2,
                             unsigned int* __restrict__ a_tok,
                             float* __restrict__ a_gate,
                             unsigned int* __restrict__ counts) {
  const int e = blockIdx.x;
  const int lane = threadIdx.x;
  unsigned int cnt = 0;
  for (int base = 0; base < NTOK; base += 32) {
    int t = base + lane;
    unsigned int tp = tops[t];
    int i0 = (int)(tp & 0xFFFFu), i1 = (int)(tp >> 16);
    bool m0 = (i0 == e), m1 = (i1 == e);
    bool match = m0 || m1;
    float g = 0.0f;
    if (match) { float2 gg = gates2[t]; g = m0 ? gg.x : gg.y; }
    unsigned int mask = (unsigned int)__ballot(match);
    if (match) {
      unsigned int off = cnt + (unsigned)__popc(mask & ((1u << lane) - 1u));
      a_tok[e * NTOK + off]  = (unsigned)t;
      a_gate[e * NTOK + off] = g;
    }
    cnt += (unsigned)__popc(mask);
  }
  if (lane == 0) counts[e] = cnt;
}

// ---------------------------------------------------------------------------
// Fused expert MLP: block = 16 tokens of one expert; 8 waves x 128 out cols.
// bf16 WMMA (16x16x32) with f32 accumulation; LDS-staged x and h chunks.
// ---------------------------------------------------------------------------
__global__ __launch_bounds__(256, 2)
void moe_expert_kernel(const unsigned short* __restrict__ xb,
                       const unsigned short* __restrict__ w1t,   // [E][H][D] bf16
                       const unsigned short* __restrict__ w2t,   // [E][D][H] bf16
                       const float* __restrict__ b1,
                       const float* __restrict__ b2,
                       const unsigned int* __restrict__ a_tok,
                       const float* __restrict__ a_gate,
                       const unsigned int* __restrict__ counts,
                       float* __restrict__ out) {
  constexpr int XP = DIMD + 8;   // 1032 elems = 129 uint4 (pad: kill bank conflicts)
  constexpr int HP = 128 + 8;    // 136 elems  = 17 uint4
  __shared__ __align__(16) unsigned short xs[16 * XP];
  __shared__ __align__(16) unsigned short hs[16 * HP];
  __shared__ unsigned int s_tok[16];
  __shared__ float s_gate[16];

  const int e   = blockIdx.y;
  const int cnt = (int)counts[e];
  const int t0  = blockIdx.x * 16;
  if (t0 >= cnt) return;

  const int tid  = threadIdx.x;
  const int wave = tid >> 5;
  const int lane = tid & 31;
  const int r    = lane & 15;
  const int half = lane >> 4;

  if (tid < 16) {
    int srcrow = (t0 + tid < cnt) ? (t0 + tid) : t0;
    s_tok[tid]  = a_tok[e * NTOK + srcrow];
    s_gate[tid] = (t0 + tid < cnt) ? a_gate[e * NTOK + srcrow] : 0.0f;
  }
  __syncthreads();

  // stage 16 bf16 token rows into LDS (padded rows)
#pragma unroll
  for (int j = 0; j < 8; ++j) {
    int i   = tid + 256 * j;          // 0..2047
    int row = i >> 7;
    int c4  = i & 127;
    uint4 v = ((const uint4*)(xb + (size_t)s_tok[row] * DIMD))[c4];
    ((uint4*)xs)[row * (XP / 8) + c4] = v;
  }
  __syncthreads();

  const unsigned short* w1e = w1t + (size_t)e * HID * DIMD;
  const unsigned short* w2e = w2t + (size_t)e * DIMD * HID;

  v8f acc[8];
#pragma unroll
  for (int i = 0; i < 8; ++i) acc[i] = v8f{0.f,0.f,0.f,0.f,0.f,0.f,0.f,0.f};

  for (int hc = 0; hc < HID; hc += 128) {
    // ---- stage 1: h[16,128] = relu(x @ W1chunk + b1); wave owns 16 cols ----
    v8f hacc = v8f{0.f,0.f,0.f,0.f,0.f,0.f,0.f,0.f};
    const int ncol = hc + wave * 16 + r;
    const unsigned short* bbase = w1e + (size_t)ncol * DIMD;
    const unsigned short* abase = xs + r * XP;
    __builtin_prefetch(bbase + 128 * DIMD, 0, 1);   // next hidden chunk's W1
#pragma unroll 8
    for (int k0 = 0; k0 < DIMD; k0 += 32) {
      FragAB a, b;
      a.u[0] = *(const uint4*)(abase + k0 + half * 8);
      a.u[1] = *(const uint4*)(abase + k0 + 16 + half * 8);
      const unsigned short* bp = bbase + k0 + half * 8;
      b.u[0] = *(const uint4*)(bp);
      b.u[1] = *(const uint4*)(bp + 16);
      hacc = bf_wmma(a, b, hacc);
    }
    const float b1v = b1[e * HID + ncol];
    const int c = wave * 16 + r;
#pragma unroll
    for (int j = 0; j < 8; ++j) {
      float hv = fmaxf(hacc[j] + b1v, 0.0f);
      int m = j + 8 * half;            // C layout: row = j + 8*half, col = r
      hs[m * HP + c] = f2bf(hv);
    }
    __syncthreads();

    // ---- stage 2: Y[:, wave*128..+128) += h @ W2chunk ----
    const unsigned short* ab2 = hs + r * HP;
#pragma unroll
    for (int kk = 0; kk < 128; kk += 32) {
      FragAB a;
      a.u[0] = *(const uint4*)(ab2 + kk + half * 8);
      a.u[1] = *(const uint4*)(ab2 + kk + 16 + half * 8);
      const int kg = hc + kk + half * 8;
#pragma unroll
      for (int nt = 0; nt < 8; ++nt) {
        int n = wave * 128 + nt * 16 + r;
        const unsigned short* bp = w2e + (size_t)n * HID + kg;
        FragAB b;
        b.u[0] = *(const uint4*)(bp);
        b.u[1] = *(const uint4*)(bp + 16);
        acc[nt] = bf_wmma(a, b, acc[nt]);
      }
    }
    __syncthreads();
  }

  // epilogue: +b2, scale by gate, scatter-add into output (2 adds/elem total)
#pragma unroll
  for (int nt = 0; nt < 8; ++nt) {
    const int d = wave * 128 + nt * 16 + r;
    const float b2v = b2[e * DIMD + d];
#pragma unroll
    for (int j = 0; j < 8; ++j) {
      int m = j + 8 * half;
      float val = (acc[nt][j] + b2v) * s_gate[m];
      atomicAdd(out + (size_t)s_tok[m] * DIMD + d, val);
    }
  }
}

// ---------------------------------------------------------------------------
// Aux losses
// ---------------------------------------------------------------------------
__global__ void finalize_kernel(const float* __restrict__ stats, float* __restrict__ out) {
  float ent = stats[0] / (float)NTOK;
  float lbl = 0.0f;
#pragma unroll
  for (int e = 0; e < 8; ++e) {
    float imp = stats[1 + e] / (float)NTOK;
    float d = imp - 0.125f;
    lbl += d * d;
  }
  lbl *= (1.0f / 8.0f);
  out[(size_t)NTOK * DIMD]     = ent;
  out[(size_t)NTOK * DIMD + 1] = lbl;
}

// ---------------------------------------------------------------------------
extern "C" void kernel_launch(void* const* d_in, const int* in_sizes, int n_in,
                              void* d_out, int out_size, void* d_ws, size_t ws_size,
                              hipStream_t stream) {
  const float* x  = (const float*)d_in[0];
  const float* rw = (const float*)d_in[1];
  const float* rb = (const float*)d_in[2];
  const float* w1 = (const float*)d_in[3];
  const float* b1 = (const float*)d_in[4];
  const float* w2 = (const float*)d_in[5];
  const float* b2 = (const float*)d_in[6];
  float* out = (float*)d_out;

  char* ws = (char*)d_ws;
  size_t off = 0;
  unsigned short* w1t = (unsigned short*)(ws + off); off += (size_t)NEXP * HID * DIMD * 2;
  unsigned short* w2t = (unsigned short*)(ws + off); off += (size_t)NEXP * DIMD * HID * 2;
  unsigned short* xb  = (unsigned short*)(ws + off); off += (size_t)NTOK * DIMD * 2;
  unsigned int* a_tok = (unsigned int*)(ws + off);   off += (size_t)NEXP * NTOK * 4;
  float* a_gate       = (float*)(ws + off);          off += (size_t)NEXP * NTOK * 4;
  unsigned int* counts= (unsigned int*)(ws + off);   off += 64;
  float* stats        = (float*)(ws + off);          off += 64;
  unsigned int* tops  = (unsigned int*)(ws + off);   off += (size_t)NTOK * 4;
  float2* gates2      = (float2*)(ws + off);         off += (size_t)NTOK * 8;

  const int n_out = NTOK * DIMD + 2;
  zero_kernel<<<(n_out + 255) / 256, 256, 0, stream>>>(out, n_out, stats);
  convx_kernel<<<(NTOK * DIMD / 4) / 256, 256, 0, stream>>>(x, xb);
  dim3 tb(32, 8);
  transpose_kernel<<<dim3(HID / 32, DIMD / 32, NEXP), tb, 0, stream>>>(w1, w1t, DIMD, HID);
  transpose_kernel<<<dim3(DIMD / 32, HID / 32, NEXP), tb, 0, stream>>>(w2, w2t, HID, DIMD);
  router_kernel<<<NTOK / 256, 256, 0, stream>>>(x, rw, rb, tops, gates2, stats);
  build_kernel<<<NEXP, 32, 0, stream>>>(tops, gates2, a_tok, a_gate, counts);
  moe_expert_kernel<<<dim3(NTOK / 16, NEXP), 256, 0, stream>>>(xb, w1t, w2t, b1, b2,
                                                               a_tok, a_gate, counts, out);
  finalize_kernel<<<1, 1, 0, stream>>>(stats, out);
}
